// PSN_64682207478109
// MI455X (gfx1250) — matrix operand order repaired and checked by reading, not compiled, not run on hardware
//
#include <hip/hip_runtime.h>
#include <math.h>

// ---------------- problem constants ----------------
#define NROWS 8192      // B*E = 32*256 latent rows
#define CDIM  256       // channel dim
#define ODIM  8192      // codebook entries
#define NTOT  2097152   // NROWS*CDIM total elements
#define BETA  0.25f

// LDS tile geometry (rows padded 256 -> 264 ushorts for bank-conflict-free ds_load_b128)
#define LROW   264                      // padded row stride in ushorts (528 B)
#define LTILE  (16 * LROW)              // one 16-row bf16 tile in LDS: 4224 ushorts (8448 B)
#define SMEM_TILES_US (LTILE * 8)       // 2 cols x 2 bufs x (hi+lo) = 8 tiles = 33792 ushorts
#define SMEM_BYTES (SMEM_TILES_US * 2 + 128 * 4 + 128 * 4)   // + red values + red indices

// ---------------- trivial vector types ----------------
typedef __attribute__((ext_vector_type(4)))  unsigned int u32x4;
typedef __attribute__((ext_vector_type(4)))  float        f32x4;
typedef __attribute__((ext_vector_type(16))) __bf16       v16bf;
typedef __attribute__((ext_vector_type(8)))  float        v8f;

union Frag { u32x4 q[2]; v16bf v; };   // 32 bytes = 8 VGPRs = one WMMA 16-bit operand

// ---------------- 1) split fp32 -> bf16 hi + bf16 residual ----------------
__device__ __forceinline__ unsigned short bf16_rne(float f) {
    unsigned int u = __float_as_uint(f);
    unsigned int r = (u + 0x7fffu + ((u >> 16) & 1u)) >> 16;   // round-to-nearest-even
    return (unsigned short)r;
}

__global__ void prep_split_kernel(const float* __restrict__ x,
                                  const float* __restrict__ cb,
                                  unsigned short* __restrict__ xh,
                                  unsigned short* __restrict__ xl,
                                  unsigned short* __restrict__ ch,
                                  unsigned short* __restrict__ cl) {
    for (int i = blockIdx.x * blockDim.x + threadIdx.x; i < 2 * NTOT;
         i += gridDim.x * blockDim.x) {
        const bool isx = (i < NTOT);
        const int  j   = isx ? i : i - NTOT;
        float v  = isx ? x[j] : cb[j];
        unsigned short h = bf16_rne(v);
        float hf = __uint_as_float(((unsigned int)h) << 16);
        unsigned short l = bf16_rne(v - hf);
        if (isx) { xh[j] = h; xl[j] = l; } else { ch[j] = h; cl[j] = l; }
    }
}

// ---------------- 2) ||codebook_o||^2 in fp32 ----------------
__global__ void cnorm_kernel(const float* __restrict__ cb, float* __restrict__ cnorm) {
    int row = blockIdx.x * blockDim.x + threadIdx.x;
    if (row >= ODIM) return;
    const f32x4* p = (const f32x4*)(cb + (size_t)row * CDIM);
    float s = 0.f;
    #pragma unroll 8
    for (int i = 0; i < CDIM / 4; ++i) {
        f32x4 v = p[i];
        s += v.x * v.x + v.y * v.y + v.z * v.z + v.w * v.w;
    }
    cnorm[row] = s;
}

// ---------------- 3) fused WMMA distance-GEMM + argmin ----------------
// grid = 128 blocks, block = 256 threads (8 waves).
// Block b owns x-rows [64b, 64b+64). Wave w: mrow = w&3 (16 rows), col = w>>2 (N half).
// Column c scans tiles n0 = (c*256 + iter)*16, iter = 0..255.
// B tiles (codebook hi+lo) are staged global->LDS with async-copy, double buffered,
// shared by the 4 waves of each column.  score = ||c||^2 - 2*x.c
__global__ __launch_bounds__(256, 1) void argmin_kernel(
    const unsigned short* __restrict__ xh, const unsigned short* __restrict__ xl,
    const unsigned short* __restrict__ ch, const unsigned short* __restrict__ cl,
    const float* __restrict__ cnorm, int* __restrict__ inds_det) {

    extern __shared__ unsigned short smem_us[];
    float* redv = (float*)(smem_us + SMEM_TILES_US);
    int*   redi = (int*)(redv + 128);

    const int tid  = threadIdx.x;
    const int lane = tid & 31;
    const int wave = tid >> 5;
    const int half = lane >> 4;       // 0: rows M0-7 / B K0-7,16-23 ; 1: rows M8-15 / B K8-15,24-31
    const int lr   = lane & 15;       // A row-in-tile ; B col-in-tile ; C/D N column
    const int mrow = wave & 3;        // M sub-block within this workgroup
    const int col  = wave >> 2;       // which N half this wave scans
    const int rowbase = blockIdx.x * 64 + mrow * 16;
    const unsigned smem_base = (unsigned)(size_t)(void*)smem_us;

    // ---- A fragments (x rows, hi+lo) resident in registers across all tiles ----
    Frag aH[8], aL[8];
    {
        const unsigned short* bh = xh + (size_t)(rowbase + lr) * CDIM + half * 8;
        const unsigned short* bl = xl + (size_t)(rowbase + lr) * CDIM + half * 8;
        #pragma unroll
        for (int kk = 0; kk < 8; ++kk) {
            aH[kk].q[0] = *(const u32x4*)(bh + kk * 32);
            aH[kk].q[1] = *(const u32x4*)(bh + kk * 32 + 16);
            aL[kk].q[0] = *(const u32x4*)(bl + kk * 32);
            aL[kk].q[1] = *(const u32x4*)(bl + kk * 32 + 16);
        }
    }

    // ---- async staging of one iteration's two B tiles (hi+lo, both columns) ----
    // 2 tiles * 16KB = 2048 x 16B chunks; 8 chunks per thread.
    // chunk c: col = c>>10, arr(hi/lo) = (c>>9)&1, row = (c>>5)&15, off = c&31
    auto stage = [&](int iter, int pbuf) {
        #pragma unroll
        for (int r = 0; r < 8; ++r) {
            int c    = tid + r * 256;
            int ccol = c >> 10;
            int arr  = (c >> 9) & 1;
            int row  = (c >> 5) & 15;
            int off  = c & 31;                 // 16B chunk within the 512B row
            const unsigned short* src =
                (arr ? cl : ch) + (size_t)((ccol * 256 + iter) * 16 + row) * CDIM + off * 8;
            unsigned lds_addr = smem_base +
                (unsigned)((((ccol * 2 + pbuf) * 2 + arr) * LTILE + row * LROW + off * 8) * 2);
            asm volatile("global_load_async_to_lds_b128 %0, %1, off"
                         :: "v"(lds_addr), "v"((unsigned long long)(size_t)src)
                         : "memory");
        }
    };

    stage(0, 0);                                   // prefetch first tiles
    asm volatile("s_wait_asynccnt 0x0" ::: "memory");
    __syncthreads();

    float best[8];
    int   bidx[8];
    #pragma unroll
    for (int i = 0; i < 8; ++i) { best[i] = 3.4e38f; bidx[i] = 0x7fffffff; }

    for (int iter = 0; iter < 256; ++iter) {
        const int pbuf = iter & 1;
        if (iter + 1 < 256) stage(iter + 1, pbuf ^ 1);   // overlap copy with WMMAs

        // B fragment base inside LDS for this wave's column, this buffer.
        // B 32x16 layout: lane<16 = col lr K[0..7],K[16..23]; lane>=16 = col lr K[8..15],K[24..31]
        const unsigned short* tb = smem_us + ((col * 2 + pbuf) * 2) * LTILE;
        const unsigned short* bbh = tb + lr * LROW + half * 8;            // hi tile
        const unsigned short* bbl = tb + LTILE + lr * LROW + half * 8;    // lo tile

        v8f acc = {};
        #pragma unroll
        for (int kk = 0; kk < 8; ++kk) {
            Frag bH, bL;
            bH.q[0] = *(const u32x4*)(bbh + kk * 32);
            bH.q[1] = *(const u32x4*)(bbh + kk * 32 + 16);
            bL.q[0] = *(const u32x4*)(bbl + kk * 32);
            bL.q[1] = *(const u32x4*)(bbl + kk * 32 + 16);
            // split-bf16: x.c ~= xh.ch + xh.cl + xl.ch  (fp32 accumulate)
            acc = __builtin_amdgcn_wmma_f32_16x16x32_bf16(false, aH[kk].v, false, bH.v,
                                                          (short)0, acc, false, false);
            acc = __builtin_amdgcn_wmma_f32_16x16x32_bf16(false, aH[kk].v, false, bL.v,
                                                          (short)0, acc, false, false);
            acc = __builtin_amdgcn_wmma_f32_16x16x32_bf16(false, aL[kk].v, false, bH.v,
                                                          (short)0, acc, false, false);
        }

        const int   n  = (col * 256 + iter) * 16 + lr;
        const float cn = cnorm[n];
        #pragma unroll
        for (int i = 0; i < 8; ++i) {          // C/D: VGPR i = row half*8+i, lane = col n
            float s = cn - 2.0f * acc[i];
            if (s < best[i]) { best[i] = s; bidx[i] = n; }   // strict <: first-min tie-break
        }

        asm volatile("s_wait_asynccnt 0x0" ::: "memory");    // tile iter+1 landed
        __syncthreads();                                     // visible to all waves
    }

    // ---- reduce over the 16 lanes of each half (same rows, different N columns) ----
    #pragma unroll
    for (int i = 0; i < 8; ++i) {
        float b  = best[i];
        int   ix = bidx[i];
        #pragma unroll
        for (int m = 8; m >= 1; m >>= 1) {
            float ob = __shfl_xor(b, m, 32);
            int   oi = __shfl_xor(ix, m, 32);
            if (ob < b || (ob == b && oi < ix)) { b = ob; ix = oi; }
        }
        if (lr == 0) {
            int r = mrow * 16 + half * 8 + i;
            redv[col * 64 + r] = b;
            redi[col * 64 + r] = ix;
        }
    }
    __syncthreads();

    // ---- combine the two N-column candidates per row ----
    if (tid < 64) {
        float b0 = redv[tid];      int i0 = redi[tid];
        float b1 = redv[64 + tid]; int i1 = redi[64 + tid];
        int ix = (b1 < b0 || (b1 == b0 && i1 < i0)) ? i1 : i0;
        inds_det[blockIdx.x * 64 + tid] = ix;
    }
}

// ---------------- 4) JAX threefry2x32 noise -> noisy indices ----------------
__device__ __forceinline__ unsigned int rotl32(unsigned int v, int r) {
    return (v << r) | (v >> (32 - r));
}

__global__ void noise_kernel(const int* __restrict__ inds_det,
                             const void* __restrict__ std_raw,
                             int* __restrict__ inds_noisy) {
    // scalar may arrive as int32 or float32; disambiguate from raw bits
    unsigned int sb = *(const unsigned int*)std_raw;
    float stdv = (sb < 0x01000000u) ? (float)(int)sb : __uint_as_float(sb);

    const unsigned int ks0 = 0u, ks1 = 42u;                 // jax.random.key(42)
    const unsigned int ks2 = ks0 ^ ks1 ^ 0x1BD11BDAu;
    const float lo = -0.99999994f;                          // nextafter(-1, 0)

    for (int i = threadIdx.x; i < NROWS / 2; i += blockDim.x) {
        unsigned int x0 = (unsigned int)i + ks0;
        unsigned int x1 = (unsigned int)(i + NROWS / 2) + ks1;
        #define TFR(r) { x0 += x1; x1 = rotl32(x1, r); x1 ^= x0; }
        TFR(13) TFR(15) TFR(26) TFR(6)  x0 += ks1; x1 += ks2 + 1u;
        TFR(17) TFR(29) TFR(16) TFR(24) x0 += ks2; x1 += ks0 + 2u;
        TFR(13) TFR(15) TFR(26) TFR(6)  x0 += ks0; x1 += ks1 + 3u;
        TFR(17) TFR(29) TFR(16) TFR(24) x0 += ks1; x1 += ks2 + 4u;
        TFR(13) TFR(15) TFR(26) TFR(6)  x0 += ks2; x1 += ks0 + 5u;
        #undef TFR

        unsigned int bits[2] = { x0, x1 };
        int outidx[2] = { i, i + NROWS / 2 };
        #pragma unroll
        for (int k = 0; k < 2; ++k) {
            float f = __uint_as_float((bits[k] >> 9) | 0x3f800000u) - 1.0f;  // [0,1)
            float u = f * (1.0f - lo) + lo;
            u = fmaxf(lo, u);
            float nrm = 1.41421356237309515f * erfinvf(u);   // jax.random.normal
            int noise = (int)rintf(stdv * nrm);              // jnp.round half-to-even
            int v = inds_det[outidx[k]] + noise;
            v = min(max(v, 0), ODIM - 1);
            inds_noisy[outidx[k]] = v;
        }
    }
}

// ---------------- 5) gather + out + per-row loss partials ----------------
__global__ __launch_bounds__(256) void out_loss_kernel(
    const float* __restrict__ x, const float* __restrict__ y,
    const float* __restrict__ cb,
    const int* __restrict__ inds_det, const int* __restrict__ inds_noisy,
    float* __restrict__ out, float* __restrict__ partials) {

    __shared__ float sd[256];
    const int row = blockIdx.x;
    const int c   = threadIdx.x;
    const int jd = inds_det[row];
    const int jn = inds_noisy[row];
    const size_t base = (size_t)row * CDIM + c;

    float qd = cb[(size_t)jd * CDIM + c];
    float qn = cb[(size_t)jn * CDIM + c];
    float xv = x[base], yv = y[base];

    out[base] = qn;                           // out = x + sg(q_noisy - x) == q_noisy (values)

    float d1 = qn - yv;                       // mse(out, y)
    float d2 = xv - qd;                       // beta * mse(x, sg(q_det))
    float d3 = qn - xv;                       // mse(q_noisy, sg(x))
    sd[c] = d1 * d1 + BETA * d2 * d2 + d3 * d3;
    __syncthreads();
    for (int s = 128; s > 0; s >>= 1) {
        if (c < s) sd[c] += sd[c + s];
        __syncthreads();
    }
    if (c == 0) partials[row] = sd[0];
}

// ---------------- 6) deterministic final reduction ----------------
__global__ void final_loss_kernel(const float* __restrict__ partials,
                                  float* __restrict__ loss_out) {
    __shared__ float sd[256];
    float s = 0.f;
    for (int i = threadIdx.x; i < NROWS; i += 256) s += partials[i];
    sd[threadIdx.x] = s;
    __syncthreads();
    for (int st = 128; st > 0; st >>= 1) {
        if (threadIdx.x < st) sd[threadIdx.x] += sd[threadIdx.x + st];
        __syncthreads();
    }
    if (threadIdx.x == 0) *loss_out = sd[0] * (1.0f / (float)NTOT);
}

// ---------------- launcher ----------------
extern "C" void kernel_launch(void* const* d_in, const int* in_sizes, int n_in,
                              void* d_out, int out_size, void* d_ws, size_t ws_size,
                              hipStream_t stream) {
    const float* x    = (const float*)d_in[0];
    const float* y    = (const float*)d_in[1];
    const float* cb   = (const float*)d_in[2];
    const void*  stdp = d_in[3];
    float* out = (float*)d_out;

    // workspace layout (~16.2 MB)
    char* ws = (char*)d_ws;
    unsigned short* xh = (unsigned short*)(ws + 0);
    unsigned short* xl = (unsigned short*)(ws + 4194304);
    unsigned short* ch = (unsigned short*)(ws + 8388608);
    unsigned short* cl = (unsigned short*)(ws + 12582912);
    float* cnorm       = (float*)(ws + 16777216);
    int*   inds_det    = (int*)  (ws + 16777216 + 32768);
    int*   inds_noisy  = (int*)  (ws + 16777216 + 65536);
    float* partials    = (float*)(ws + 16777216 + 98304);

    prep_split_kernel<<<2048, 256, 0, stream>>>(x, cb, xh, xl, ch, cl);
    cnorm_kernel<<<ODIM / 256, 256, 0, stream>>>(cb, cnorm);
    argmin_kernel<<<NROWS / 64, 256, SMEM_BYTES, stream>>>(xh, xl, ch, cl, cnorm, inds_det);
    noise_kernel<<<1, 256, 0, stream>>>(inds_det, stdp, inds_noisy);
    out_loss_kernel<<<NROWS, 256, 0, stream>>>(x, y, cb, inds_det, inds_noisy, out, partials);
    final_loss_kernel<<<1, 256, 0, stream>>>(partials, out + (size_t)NTOT);
}
